// PrototypeNet_9182640079479
// MI455X (gfx1250) — matrix operand order repaired and checked by reading, not compiled, not run on hardware
//
#include <hip/hip_runtime.h>
#include <hip/hip_bf16.h>

typedef __attribute__((ext_vector_type(16))) _Float16 v16h;
typedef __attribute__((ext_vector_type(8)))  _Float16 v8h;
typedef __attribute__((ext_vector_type(8)))  float    v8f;
typedef _Float16 f16;

#define EPSV 1e-5f

// Async global->LDS copy support (gfx1250). Guarded so we fall back to a
// plain VALU staging path if the builtin is absent in this toolchain.
#define HAVE_ASYNC_LDS 0
#if defined(__has_builtin)
#if __has_builtin(__builtin_amdgcn_global_load_async_to_lds_b128)
#undef HAVE_ASYNC_LDS
#define HAVE_ASYNC_LDS 1
#endif
#endif

#if HAVE_ASYNC_LDS
// signature (from compiler diagnostic): (global int4*, shared int4*, imm, imm)
typedef int v4i_async __attribute__((vector_size(16)));
__device__ __forceinline__ void async_copy16(const void* g, void* l) {
  __builtin_amdgcn_global_load_async_to_lds_b128(
      (__attribute__((address_space(1))) v4i_async*)g,
      (__attribute__((address_space(3))) v4i_async*)l, 0, 0);
}
#endif

// -------------------------------------------------------------------------
// small utility kernels
// -------------------------------------------------------------------------
__global__ void zero_stats_kernel(float* sum, float* sqs, int n) {
  int i = blockIdx.x * blockDim.x + threadIdx.x;
  if (i < n) { sum[i] = 0.f; sqs[i] = 0.f; }
}

// pure fp32 -> f16 convert, keeps OIHW ([Cout][K]) layout
__global__ void wconvert_kernel(const float* __restrict__ W, f16* __restrict__ Wh,
                                int total) {
  int i = blockIdx.x * blockDim.x + threadIdx.x;
  if (i < total) Wh[i] = (f16)W[i];
}

// fold batch stats + gamma/beta into per-channel scale/shift
__global__ void stats_kernel(const float* __restrict__ sum, const float* __restrict__ sqs,
                             const float* __restrict__ g, const float* __restrict__ beta,
                             float* __restrict__ scale, float* __restrict__ shift,
                             int C, float invM) {
  int c = blockIdx.x * blockDim.x + threadIdx.x;
  if (c < C) {
    float mu  = sum[c] * invM;
    float var = sqs[c] * invM - mu * mu;
    float sc  = g[c] * rsqrtf(var + EPSV);
    scale[c] = sc;
    shift[c] = beta[c] - mu * sc;
  }
}

// normalize + ReLU + convert to f16 (NCHW); HW and C are powers of two
__global__ void bnrelu_kernel(const float* __restrict__ x,
                              const float* __restrict__ scale, const float* __restrict__ shift,
                              f16* __restrict__ y, int logHW, int Cmask, long total) {
  long i = (long)blockIdx.x * blockDim.x + threadIdx.x;
  if (i < total) {
    int c = (int)(i >> logHW) & Cmask;
    float v = x[i] * scale[c] + shift[c];
    y[i] = (f16)fmaxf(v, 0.f);
  }
}

// -------------------------------------------------------------------------
// block 1: direct conv, Cin=1, 128x128 -> 64x64 stride 2, Cout=32 (fp32)
// -------------------------------------------------------------------------
__global__ void conv1_kernel(const float* __restrict__ xs, const float* __restrict__ xq,
                             const float* __restrict__ W, const float* __restrict__ B,
                             float* __restrict__ out, float* __restrict__ sum,
                             float* __restrict__ sqs) {
  const int HW = 64 * 64;
  const int M  = 200 * HW;
  int c = blockIdx.y;
  int p = blockIdx.x * blockDim.x + threadIdx.x;
  __shared__ float sS, sQ;
  if (threadIdx.x == 0) { sS = 0.f; sQ = 0.f; }
  __syncthreads();
  if (p < M) {
    int img = p / HW, rem = p % HW, oy = rem >> 6, ox = rem & 63;
    const float* src = (img < 100) ? (xs + (long)img * 128 * 128)
                                   : (xq + (long)(img - 100) * 128 * 128);
    float acc = B[c];
    #pragma unroll
    for (int r = 0; r < 3; ++r) {
      int iy = oy * 2 - 1 + r;
      if (iy >= 0 && iy < 128) {
        #pragma unroll
        for (int s = 0; s < 3; ++s) {
          int ix = ox * 2 - 1 + s;
          if (ix >= 0 && ix < 128)
            acc += W[c * 9 + r * 3 + s] * src[iy * 128 + ix];
        }
      }
    }
    out[((long)img * 32 + c) * HW + rem] = acc;
    atomicAdd(&sS, acc);
    atomicAdd(&sQ, acc * acc);
  }
  __syncthreads();
  if (threadIdx.x == 0) { atomicAdd(&sum[c], sS); atomicAdd(&sqs[c], sQ); }
}

// -------------------------------------------------------------------------
// blocks 2-4: implicit-GEMM conv via V_WMMA_F32_16X16X32_F16
//   workgroup = 8 waves, tile 64(M) x 32(N); wave tile 16x16; K step 32.
//   A: thread owns one K column (decode once/step) + 8 pixel rows (hoisted);
//      branchless halo (v_med3 clamp + cndmask); loads batched before stores;
//      all offsets 32-bit so loads use SADDR + 32-bit voffset (GVS mode).
//   B: LDS tile is [cout][k]; staged with GLOBAL_LOAD_ASYNC_TO_LDS_B128.
//   Fragments load as 2x ds_load_b128 each (16B-aligned 80B LDS rows).
// -------------------------------------------------------------------------
template <int Cin, int Hin, int Win, int Cout, int Hout, int Wout, int STR>
__global__ __launch_bounds__(256)
void convgemm_kernel(const f16* __restrict__ act, const f16* __restrict__ Wh,
                     const float* __restrict__ bias,
                     float* __restrict__ out, float* __restrict__ sum,
                     float* __restrict__ sqs) {
  constexpr int K   = Cin * 9;
  constexpr int HWo = Hout * Wout;
  constexpr int HWi = Hin * Win;
  __shared__ __align__(16) f16 As[64][40];   // [pixel-row][k], 80B rows
  __shared__ __align__(16) f16 Bs[32][40];   // [cout][k], 80B rows
  __shared__ float sS[32], sQ[32];

  const int tid  = threadIdx.x;
  const int lane = tid & 31;
  const int wave = tid >> 5;
  const int wm   = wave & 3;
  const int wn   = wave >> 2;
  const int tileM = blockIdx.x * 64;
  const int tileN = blockIdx.y * 32;

  if (tid < 32) { sS[tid] = 0.f; sQ[tid] = 0.f; }

  // ---- A staging geometry: one K column, 8 pixel rows per thread ----
  const int col     = tid & 31;        // K column within tile
  const int rowbase = (tid >> 5) * 8;  // 8 consecutive pixel rows
  int gb[8];    // per-row image base offset (elements; all buffers < 2^31)
  int iyA[8];   // per-row input y origin (oy*STR - 1)
  int ixA[8];   // per-row input x origin (ox*STR - 1)
  #pragma unroll
  for (int e = 0; e < 8; ++e) {
    int p   = tileM + rowbase + e;
    int img = p / HWo;                 // constants -> shifts
    int rem = p % HWo;
    int oy  = rem / Wout;
    int ox  = rem % Wout;
    iyA[e] = oy * STR - 1;
    ixA[e] = ox * STR - 1;
    gb[e]  = img * (Cin * HWi);
  }
  // incremental decode of k = kk + col -> (ci, tt); 32 = 3*9 + 5
  int ci = col / 9;
  int tt = col % 9;

  v8f acc = {};

  for (int kk = 0; kk < K; kk += 32) {
    __syncthreads();

    // ---- stage A tile: branchless; batch all loads, then all stores ----
    {
      int r = tt / 3;
      int s = tt - r * 3;
      int koff = ci * HWi;
      f16  vals[8];
      bool oks[8];
      #pragma unroll
      for (int e = 0; e < 8; ++e) {
        int iy = iyA[e] + r;
        int ix = ixA[e] + s;
        oks[e] = ((unsigned)iy < (unsigned)Hin) & ((unsigned)ix < (unsigned)Win);
        int iyc = iy < 0 ? 0 : (iy > Hin - 1 ? Hin - 1 : iy);   // v_med3
        int ixc = ix < 0 ? 0 : (ix > Win - 1 ? Win - 1 : ix);   // v_med3
        unsigned off = (unsigned)(gb[e] + koff + iyc * Win + ixc);
        vals[e] = act[off];                                     // always valid
      }
      #pragma unroll
      for (int e = 0; e < 8; ++e)
        As[rowbase + e][col] = oks[e] ? vals[e] : (f16)0.f;
    }
    ci += 3; tt += 5;
    if (tt >= 9) { tt -= 9; ci += 1; }
    if (kk + 32 < K)  // hint next activation line into cache
      __builtin_prefetch(act + (unsigned)(gb[0] + ci * HWi), 0, 0);

    // ---- stage B tile: Bs[cout][k], 32x32 f16 ----
#if HAVE_ASYNC_LDS
    if (tid < 128) {                    // 128 lanes x 16B = 2048B tile
      int nc = tid >> 2, ch = tid & 3;  // 8 consecutive k of one cout per lane
      const f16* g = Wh + (unsigned)((tileN + nc) * K + kk + ch * 8);
      async_copy16((const void*)g, (void*)&Bs[nc][ch * 8]);
    }
#if __has_builtin(__builtin_amdgcn_s_wait_asynccnt)
    __builtin_amdgcn_s_wait_asynccnt(0);
#else
    asm volatile("s_wait_asynccnt 0x0" ::: "memory");
#endif
#else
    {
      int nc = tid >> 3, k4 = (tid & 7) * 4;
      #pragma unroll
      for (int e = 0; e < 4; ++e)
        Bs[nc][k4 + e] = Wh[(unsigned)((tileN + nc) * K + kk + k4 + e)];
    }
#endif
    __syncthreads();

    // ---- build fragments: two 16B runs per lane (k 8h+s, 16+8h+s) -> 2x ds_load_b128 ----
    v16h af, bf;
    {
      int arow = wm * 16 + (lane & 15);
      int kb   = (lane < 16) ? 0 : 8;      // A: half-wave K interleave
      v8h a0 = *(const v8h*)&As[arow][kb];
      v8h a1 = *(const v8h*)&As[arow][16 + kb];
      af = __builtin_shufflevector(a0, a1, 0, 1, 2, 3, 4, 5, 6, 7,
                                   8, 9, 10, 11, 12, 13, 14, 15);
      int ncol = wn * 16 + (lane & 15);
      int kb2  = (lane < 16) ? 0 : 8;      // B: half-wave K interleave (same as A)
      v8h b0 = *(const v8h*)&Bs[ncol][kb2];
      v8h b1 = *(const v8h*)&Bs[ncol][16 + kb2];
      bf = __builtin_shufflevector(b0, b1, 0, 1, 2, 3, 4, 5, 6, 7,
                                   8, 9, 10, 11, 12, 13, 14, 15);
    }
    acc = __builtin_amdgcn_wmma_f32_16x16x32_f16(false, af, false, bf,
                                                 (short)0, acc, false, false);
  }

  // ---- epilogue: +bias, store NCHW fp32, reduce per-channel stats ----
  int colN = lane & 15;
  int ch   = tileN + wn * 16 + colN;
  int mb   = (lane < 16) ? 0 : 8;          // D: VGPR r -> M=r (lo) / r+8 (hi)
  float b  = bias[ch];
  float lsum = 0.f, lsqs = 0.f;
  #pragma unroll
  for (int r = 0; r < 8; ++r) {
    int p = tileM + wm * 16 + mb + r;
    float v = acc[r] + b;
    int pimg = p / HWo, prem = p % HWo;    // constants -> shifts
    out[((long)pimg * Cout + ch) * HWo + prem] = v;
    lsum += v; lsqs += v * v;
  }
  atomicAdd(&sS[wn * 16 + colN], lsum);
  atomicAdd(&sQ[wn * 16 + colN], lsqs);
  __syncthreads();
  if (tid < 32) {
    atomicAdd(&sum[tileN + tid], sS[tid]);
    atomicAdd(&sqs[tileN + tid], sQ[tid]);
  }
}

// -------------------------------------------------------------------------
// head: pool -> prototypes -> distances -> softmax
// -------------------------------------------------------------------------
__global__ void pool_kernel(const f16* __restrict__ act, float* __restrict__ z) {
  // act: [200][256][16][16] f16 ; z: [200][256]
  int i = blockIdx.x * blockDim.x + threadIdx.x;  // img*256 + c
  if (i < 200 * 256) {
    const f16* p = act + (long)i * 256;
    float s = 0.f;
    for (int j = 0; j < 256; ++j) s += (float)p[j];
    z[i] = s * (1.f / 256.f);
  }
}

__global__ void proto_kernel(const float* __restrict__ z, float* __restrict__ cs) {
  int i = blockIdx.x * blockDim.x + threadIdx.x;  // j*256 + c
  if (i < 20 * 256) {
    int j = i >> 8, c = i & 255;
    float s = 0.f;
    for (int k = 0; k < 5; ++k) s += z[(j * 5 + k) * 256 + c];
    cs[i] = s * 0.2f;
  }
}

__global__ void dist_kernel(const float* __restrict__ z, const float* __restrict__ cs,
                            float* __restrict__ out) {
  int i = blockIdx.x * blockDim.x + threadIdx.x;  // q20*20 + j
  if (i < 400) {
    int q20 = i / 20, j = i % 20;
    float d[5];
    for (int q5 = 0; q5 < 5; ++q5) {
      int qi = 100 + q5 * 20 + q20;               // query image index
      const float* zq = z  + (long)qi * 256;
      const float* cj = cs + (long)j * 256;
      float s = 0.f;
      for (int c = 0; c < 256; ++c) { float t = zq[c] - cj[c]; s += t * t; }
      d[q5] = s * (1.f / 256.f);
    }
    float m = d[0];
    for (int q5 = 1; q5 < 5; ++q5) m = fmaxf(m, d[q5]);
    float e[5], es = 0.f;
    for (int q5 = 0; q5 < 5; ++q5) { e[q5] = expf(d[q5] - m); es += e[q5]; }
    for (int q5 = 0; q5 < 5; ++q5)
      out[(q20 * 5 + q5) * 20 + j] = e[q5] / es;   // [20,5,20]
  }
}

// -------------------------------------------------------------------------
// launcher
// -------------------------------------------------------------------------
extern "C" void kernel_launch(void* const* d_in, const int* in_sizes, int n_in,
                              void* d_out, int out_size, void* d_ws, size_t ws_size,
                              hipStream_t stream) {
  (void)in_sizes; (void)n_in; (void)out_size; (void)ws_size;
  char* ws = (char*)d_ws;
  float* raw   = (float*)(ws);                         // 104,857,600 B (max layer1 out)
  f16*   actA  = (f16*)(ws + 104857600);               //  52,428,800 B
  f16*   actB  = (f16*)(ws + 157286400);               //  26,214,400 B
  f16*   Wh    = (f16*)(ws + 183500800);               //     589,824 B (max weights)
  float* sum   = (float*)(ws + 184090624);
  float* sqs   = (float*)(ws + 184091648);
  float* scale = (float*)(ws + 184092672);
  float* shift = (float*)(ws + 184093696);
  float* z     = (float*)(ws + 184094720);             // 200*256 fp32
  float* cs    = (float*)(ws + 184299520);             // 20*256 fp32

  const float* xs = (const float*)d_in[0];
  const float* xq = (const float*)d_in[1];
  auto Wp = [&](int l) { return (const float*)d_in[2 + 4 * l]; };
  auto Bp = [&](int l) { return (const float*)d_in[3 + 4 * l]; };
  auto Gp = [&](int l) { return (const float*)d_in[4 + 4 * l]; };
  auto Ep = [&](int l) { return (const float*)d_in[5 + 4 * l]; };

  // ---- layer 1: direct conv (Cin=1), 128->64, Cout=32 ----
  zero_stats_kernel<<<1, 256, 0, stream>>>(sum, sqs, 256);
  conv1_kernel<<<dim3(3200, 32), 256, 0, stream>>>(xs, xq, Wp(0), Bp(0), raw, sum, sqs);
  stats_kernel<<<1, 256, 0, stream>>>(sum, sqs, Gp(0), Ep(0), scale, shift, 32,
                                      1.f / (200.f * 64 * 64));
  { long t = 200L * 32 * 64 * 64;
    bnrelu_kernel<<<(int)((t + 255) / 256), 256, 0, stream>>>(raw, scale, shift, actA,
                                                              12, 31, t); }

  // ---- layer 2: WMMA GEMM, 32->64, 64->32 stride 2 (M=204800, K=288) ----
  wconvert_kernel<<<(288 * 64 + 255) / 256, 256, 0, stream>>>(Wp(1), Wh, 288 * 64);
  zero_stats_kernel<<<1, 256, 0, stream>>>(sum, sqs, 256);
  convgemm_kernel<32, 64, 64, 64, 32, 32, 2>
      <<<dim3(204800 / 64, 64 / 32), 256, 0, stream>>>(actA, Wh, Bp(1), raw, sum, sqs);
  stats_kernel<<<1, 256, 0, stream>>>(sum, sqs, Gp(1), Ep(1), scale, shift, 64,
                                      1.f / (200.f * 32 * 32));
  { long t = 200L * 64 * 32 * 32;
    bnrelu_kernel<<<(int)((t + 255) / 256), 256, 0, stream>>>(raw, scale, shift, actB,
                                                              10, 63, t); }

  // ---- layer 3: WMMA GEMM, 64->128, 32->16 stride 2 (M=51200, K=576) ----
  wconvert_kernel<<<(576 * 128 + 255) / 256, 256, 0, stream>>>(Wp(2), Wh, 576 * 128);
  zero_stats_kernel<<<1, 256, 0, stream>>>(sum, sqs, 256);
  convgemm_kernel<64, 32, 32, 128, 16, 16, 2>
      <<<dim3(51200 / 64, 128 / 32), 256, 0, stream>>>(actB, Wh, Bp(2), raw, sum, sqs);
  stats_kernel<<<1, 256, 0, stream>>>(sum, sqs, Gp(2), Ep(2), scale, shift, 128,
                                      1.f / (200.f * 16 * 16));
  { long t = 200L * 128 * 16 * 16;
    bnrelu_kernel<<<(int)((t + 255) / 256), 256, 0, stream>>>(raw, scale, shift, actA,
                                                              8, 127, t); }

  // ---- layer 4: WMMA GEMM, 128->256, 16->16 stride 1 (M=51200, K=1152) ----
  wconvert_kernel<<<(1152 * 256 + 255) / 256, 256, 0, stream>>>(Wp(3), Wh, 1152 * 256);
  zero_stats_kernel<<<1, 256, 0, stream>>>(sum, sqs, 256);
  convgemm_kernel<128, 16, 16, 256, 16, 16, 1>
      <<<dim3(51200 / 64, 256 / 32), 256, 0, stream>>>(actA, Wh, Bp(3), raw, sum, sqs);
  stats_kernel<<<1, 256, 0, stream>>>(sum, sqs, Gp(3), Ep(3), scale, shift, 256,
                                      1.f / (200.f * 16 * 16));
  { long t = 200L * 256 * 16 * 16;
    bnrelu_kernel<<<(int)((t + 255) / 256), 256, 0, stream>>>(raw, scale, shift, actB,
                                                              8, 255, t); }

  // ---- head ----
  pool_kernel<<<200, 256, 0, stream>>>(actB, z);
  proto_kernel<<<20, 256, 0, stream>>>(z, cs);
  dist_kernel<<<2, 256, 0, stream>>>(z, cs, (float*)d_out);
}
